// san_conv_55001351193237
// MI455X (gfx1250) — matrix operand order repaired and checked by reading, not compiled, not run on hardware
//
#include <hip/hip_runtime.h>
#include <hip/hip_bf16.h>
#include <math.h>

#define NN 4096
#define FF 32
#define NSPLIT 8           // K/j-dimension splits -> 32x8 = 256 blocks per heavy kernel
#define KLEN (NN / NSPLIT) // 512 columns per split

typedef __attribute__((ext_vector_type(16))) __bf16 v16bf;
typedef __attribute__((ext_vector_type(8)))  float  v8f;

// WMMA A-operand (16x32 bf16): row = lane&15, K = g*8+i (i<8) / 16+g*8+(i-8), g = lane>>4
__device__ __forceinline__ v16bf load_a_tile(const float* __restrict__ Arow, int k, int g) {
    v16bf a;
    const float* p0 = Arow + k + g * 8;
    const float* p1 = Arow + k + 16 + g * 8;
#pragma unroll
    for (int i = 0; i < 8; ++i) { a[i] = (__bf16)p0[i]; a[8 + i] = (__bf16)p1[i]; }
    return a;
}

// WMMA B-operand (32x16 bf16) from pre-transposed bf16 Bt[n][k]:
// N = lane&15 (+16 for second tile), K = g*8+i (i<8) / 16+g*8+(i-8)
__device__ __forceinline__ v16bf load_b_tile(const __bf16* __restrict__ Btn, int k, int g) {
    v16bf b;
    const __bf16* q0 = Btn + k + g * 8;
    const __bf16* q1 = Btn + k + 16 + g * 8;
#pragma unroll
    for (int i = 0; i < 8; ++i) { b[i] = q0[i]; b[8 + i] = q1[i]; }
    return b;
}

// ---------------------------------------------------------------------------
// Kernel 1: per-row 32x32 projections + attention logits.
//   xw10/xw20 kept f32 (the +C term of M); xw11/xw21/xw0p emitted directly as
//   bf16 TRANSPOSED [32][4096] so the GEMM kernels read WMMA-ready B operands.
// ---------------------------------------------------------------------------
__global__ void k_prep(const float* __restrict__ x,
                       const float* __restrict__ W1, const float* __restrict__ W2,
                       const float* __restrict__ W0,
                       const float* __restrict__ att1, const float* __restrict__ att2,
                       float* __restrict__ xw10, float* __restrict__ xw20,
                       __bf16* __restrict__ xw11T, __bf16* __restrict__ xw21T,
                       __bf16* __restrict__ xw0pT,
                       float* __restrict__ er1, float* __restrict__ ec1,
                       float* __restrict__ er2, float* __restrict__ ec2)
{
    __shared__ float sW[5 * FF * FF];
    const int t = threadIdx.x;
    for (int i = t; i < 2 * FF * FF; i += blockDim.x) sW[i] = W1[i];
    for (int i = t; i < 2 * FF * FF; i += blockDim.x) sW[2 * FF * FF + i] = W2[i];
    for (int i = t; i < FF * FF;     i += blockDim.x) sW[4 * FF * FF + i] = W0[i];
    __syncthreads();

    const int lane = t & 31;
    const int row  = blockIdx.x * (blockDim.x >> 5) + (t >> 5);

    const float xv = x[row * FF + lane];
    float a10 = 0.f, a11 = 0.f, a20 = 0.f, a21 = 0.f, a0 = 0.f;
#pragma unroll
    for (int f = 0; f < FF; ++f) {
        const float xf = __shfl(xv, f, 32);
        a10 = fmaf(xf, sW[0 * FF * FF + f * FF + lane], a10);
        a11 = fmaf(xf, sW[1 * FF * FF + f * FF + lane], a11);
        a20 = fmaf(xf, sW[2 * FF * FF + f * FF + lane], a20);
        a21 = fmaf(xf, sW[3 * FF * FF + f * FF + lane], a21);
        a0  = fmaf(xf, sW[4 * FF * FF + f * FF + lane], a0);
    }
    xw10[row * FF + lane] = a10;
    xw20[row * FF + lane] = a20;
    xw11T[(size_t)lane * NN + row] = (__bf16)a11;
    xw21T[(size_t)lane * NN + row] = (__bf16)a21;
    xw0pT[(size_t)lane * NN + row] = (__bf16)a0;

    float er1v = a10 * att1[lane]      + a11 * att1[32 + lane];
    float ec1v = a10 * att1[64 + lane] + a11 * att1[96 + lane];
    float er2v = a20 * att2[lane]      + a21 * att2[32 + lane];
    float ec2v = a20 * att2[64 + lane] + a21 * att2[96 + lane];
#pragma unroll
    for (int o = 16; o >= 1; o >>= 1) {
        er1v += __shfl_xor(er1v, o, 32);
        ec1v += __shfl_xor(ec1v, o, 32);
        er2v += __shfl_xor(er2v, o, 32);
        ec2v += __shfl_xor(ec2v, o, 32);
    }
    if (lane == 0) {
        er1[row] = er1v;  ec1[row] = ec1v;
        er2[row] = er2v;  ec2[row] = ec2v;
    }
}

// ---------------------------------------------------------------------------
// Kernel 2: split-K partial GEMM.  Yp[split] = A[rows, ksplit] @ B[ksplit, 32].
// grid (32, NSPLIT) x 256 threads; each wave = independent 16-row tile, no LDS,
// no barriers; A streamed from HBM once total, B read from L2 (bf16, transposed).
// ---------------------------------------------------------------------------
__global__ void k_gemm_part(const float* __restrict__ A, const __bf16* __restrict__ Bt,
                            float* __restrict__ Yp)
{
    const int t = threadIdx.x, lane = t & 31, wave = t >> 5;
    const int rowbase = blockIdx.x * 128 + wave * 16;
    const int ks = blockIdx.y * KLEN;
    const int r = lane & 15, g = lane >> 4;

    const float*  Arow = A  + (size_t)(rowbase + r) * NN + ks;
    const __bf16* Bt0  = Bt + (size_t)r        * NN + ks;
    const __bf16* Bt1  = Bt + (size_t)(r + 16) * NN + ks;
    v8f acc0 = {}, acc1 = {};

    for (int kb = 0; kb < KLEN; kb += 32) {
        __builtin_prefetch(Arow + kb + 256, 0, 1);
        const v16bf a  = load_a_tile(Arow, kb, g);
        const v16bf b0 = load_b_tile(Bt0, kb, g);
        const v16bf b1 = load_b_tile(Bt1, kb, g);
        acc0 = __builtin_amdgcn_wmma_f32_16x16x32_bf16(false, a, false, b0,
                                                       (short)0, acc0, false, false);
        acc1 = __builtin_amdgcn_wmma_f32_16x16x32_bf16(false, a, false, b1,
                                                       (short)0, acc1, false, false);
    }
    float* Y = Yp + (size_t)blockIdx.y * (NN * FF);   // D: col = r(+16), row = rr+g*8
#pragma unroll
    for (int rr = 0; rr < 8; ++rr) {
        const int row = rowbase + rr + g * 8;
        Y[row * FF + r]      = acc0[rr];
        Y[row * FF + 16 + r] = acc1[rr];
    }
}

// Reduce split-K partials -> f32 row-major (base output = P @ xW0)
__global__ void k_reduce_f32(const float* __restrict__ Yp, float* __restrict__ out)
{
    const int e = blockIdx.x * 256 + threadIdx.x;
    float v = 0.f;
#pragma unroll
    for (int p = 0; p < NSPLIT; ++p) v += Yp[(size_t)p * (NN * FF) + e];
    out[e] = v;
}

// Reduce split-K partials + C  ->  bf16 transposed M_T[32][4096] (attn B operand)
__global__ void k_reduce_bf16T(const float* __restrict__ Yp, const float* __restrict__ C,
                               __bf16* __restrict__ Mt)
{
    const int e = blockIdx.x * 256 + threadIdx.x;
    float v = C[e];
#pragma unroll
    for (int p = 0; p < NSPLIT; ++p) v += Yp[(size_t)p * (NN * FF) + e];
    Mt[(size_t)(e & 31) * NN + (e >> 5)] = (__bf16)v;
}

// ---------------------------------------------------------------------------
// Kernel 3a: split-j flash-softmax partials.
//   e = leaky(er[i]+ec[j]) masked to -9e15 where L==0 (exact f32 test).
//   Emits per split: m (row max), s (row sum), acc (unnormalized, ref m).
// exp tile is built directly in WMMA A-layout; row stats live in lane=row&15,
// broadcast to D layout via __shfl for the online rescale.
// ---------------------------------------------------------------------------
__global__ void k_attn_part(const float* __restrict__ L, const float* __restrict__ er,
                            const float* __restrict__ ec, const __bf16* __restrict__ Mt,
                            float* __restrict__ accp, float* __restrict__ mp,
                            float* __restrict__ sp)
{
    const int t = threadIdx.x, lane = t & 31, wave = t >> 5;
    const int rowbase = blockIdx.x * 128 + wave * 16;
    const int js = blockIdx.y * KLEN;
    const int r = lane & 15, g = lane >> 4;

    const float*  Lrow = L  + (size_t)(rowbase + r) * NN + js;
    const float*  ecs  = ec + js;
    const __bf16* Mt0  = Mt + (size_t)r        * NN + js;
    const __bf16* Mt1  = Mt + (size_t)(r + 16) * NN + js;
    const float erv = er[rowbase + r];

    float mrow = -INFINITY, srow = 0.f;
    v8f acc0 = {}, acc1 = {};

    for (int kb = 0; kb < KLEN; kb += 32) {
        __builtin_prefetch(Lrow + kb + 256, 0, 1);
        float ev[16];
        const float* p0 = Lrow + kb + g * 8;
        const float* p1 = Lrow + kb + 16 + g * 8;
        const float* e0 = ecs + kb + g * 8;
        const float* e1 = ecs + kb + 16 + g * 8;
        float cmax = -INFINITY;
#pragma unroll
        for (int i = 0; i < 8; ++i) {
            const float l0 = p0[i], l1 = p1[i];
            float t0 = erv + e0[i]; t0 = t0 > 0.f ? t0 : 0.2f * t0;
            float t1 = erv + e1[i]; t1 = t1 > 0.f ? t1 : 0.2f * t1;
            ev[i]     = (l0 != 0.f) ? t0 : -9e15f;
            ev[8 + i] = (l1 != 0.f) ? t1 : -9e15f;
            cmax = fmaxf(cmax, fmaxf(ev[i], ev[8 + i]));
        }
        cmax = fmaxf(cmax, __shfl_xor(cmax, 16, 32));
        const float newm  = fmaxf(mrow, cmax);
        const float scale = __expf(mrow - newm);

        v16bf pa;
        float psum = 0.f;
#pragma unroll
        for (int i = 0; i < 16; ++i) {
            const float pe = __expf(ev[i] - newm);
            psum += pe;
            pa[i] = (__bf16)pe;
        }
        psum += __shfl_xor(psum, 16, 32);
        srow = srow * scale + psum;
        mrow = newm;

#pragma unroll
        for (int rr = 0; rr < 8; ++rr) {
            const float sc = __shfl(scale, rr + g * 8, 32);
            acc0[rr] *= sc;  acc1[rr] *= sc;
        }
        const v16bf b0 = load_b_tile(Mt0, kb, g);
        const v16bf b1 = load_b_tile(Mt1, kb, g);
        acc0 = __builtin_amdgcn_wmma_f32_16x16x32_bf16(false, pa, false, b0,
                                                       (short)0, acc0, false, false);
        acc1 = __builtin_amdgcn_wmma_f32_16x16x32_bf16(false, pa, false, b1,
                                                       (short)0, acc1, false, false);
    }
    float* accs = accp + (size_t)blockIdx.y * (NN * FF);
#pragma unroll
    for (int rr = 0; rr < 8; ++rr) {
        const int row = rowbase + rr + g * 8;
        accs[row * FF + r]      = acc0[rr];
        accs[row * FF + 16 + r] = acc1[rr];
    }
    if (g == 0) {   // lanes 0-15 hold the 16 row stats
        mp[blockIdx.y * NN + rowbase + r] = mrow;
        sp[blockIdx.y * NN + rowbase + r] = srow;
    }
}

// Kernel 3b: combine split-j softmax partials:  out += (sum_p acc_p*e^{m_p-M}) / S
__global__ void k_attn_reduce(const float* __restrict__ accp, const float* __restrict__ mp,
                              const float* __restrict__ sp, float* __restrict__ out)
{
    const int lane = threadIdx.x & 31;
    const int row  = blockIdx.x * 8 + (threadIdx.x >> 5);
    float M = -INFINITY;
#pragma unroll
    for (int p = 0; p < NSPLIT; ++p) M = fmaxf(M, mp[p * NN + row]);
    float S = 0.f, z = 0.f;
#pragma unroll
    for (int p = 0; p < NSPLIT; ++p) {
        const float w = __expf(mp[p * NN + row] - M);
        S += sp[p * NN + row] * w;
        z += accp[(size_t)p * (NN * FF) + row * FF + lane] * w;
    }
    out[row * FF + lane] += z / S;
}

// ---------------------------------------------------------------------------
// out = branch(Ld,W1,att1) + branch(Lu,W2,att2) + P@(x@W0)
// branch z = softmax(mask(leaky(er+ec^T),L)) @ (xw0 + L@xw1)  [associativity]
// ---------------------------------------------------------------------------
extern "C" void kernel_launch(void* const* d_in, const int* in_sizes, int n_in,
                              void* d_out, int out_size, void* d_ws, size_t ws_size,
                              hipStream_t stream) {
    (void)in_sizes; (void)n_in; (void)out_size; (void)ws_size;
    const float* x    = (const float*)d_in[0];
    const float* Ld   = (const float*)d_in[1];
    const float* Lu   = (const float*)d_in[2];
    const float* P    = (const float*)d_in[3];
    const float* W1   = (const float*)d_in[4];
    const float* W2   = (const float*)d_in[5];
    const float* W0   = (const float*)d_in[6];
    const float* att1 = (const float*)d_in[7];
    const float* att2 = (const float*)d_in[8];
    float* out = (float*)d_out;

    const size_t NF = (size_t)NN * FF;   // 131072
    float* w    = (float*)d_ws;
    float* Yp   = w;                      // NSPLIT*NF f32 (reused as attn acc partials)
    float* xw10 = w + NSPLIT * NF;
    float* xw20 = xw10 + NF;
    float* mp   = xw20 + NF;              // NSPLIT*NN
    float* sp   = mp + NSPLIT * NN;
    float* er1  = sp + NSPLIT * NN;
    float* ec1  = er1 + NN;
    float* er2  = ec1 + NN;
    float* ec2  = er2 + NN;
    __bf16* bw   = (__bf16*)(ec2 + NN);
    __bf16* xw11T = bw;                   // [32][4096] each
    __bf16* xw21T = xw11T + NF;
    __bf16* xw0pT = xw21T + NF;
    __bf16* MT    = xw0pT + NF;

    const dim3 gpart(NN / 128, NSPLIT);

    // 1) projections + logits (+ pre-transposed bf16 B operands)
    k_prep<<<NN / 8, 256, 0, stream>>>(x, W1, W2, W0, att1, att2,
                                       xw10, xw20, xw11T, xw21T, xw0pT,
                                       er1, ec1, er2, ec2);
    // 2) base: out = P @ (x@W0)
    k_gemm_part  <<<gpart,    256, 0, stream>>>(P, xw0pT, Yp);
    k_reduce_f32 <<<NF / 256, 256, 0, stream>>>(Yp, out);
    // 3) branch 1: M = xw10 + Ld@xw11 ; out += softmax(...) @ M
    k_gemm_part   <<<gpart,    256, 0, stream>>>(Ld, xw11T, Yp);
    k_reduce_bf16T<<<NF / 256, 256, 0, stream>>>(Yp, xw10, MT);
    k_attn_part   <<<gpart,    256, 0, stream>>>(Ld, er1, ec1, MT, Yp, mp, sp);
    k_attn_reduce <<<NN / 8,   256, 0, stream>>>(Yp, mp, sp, out);
    // 4) branch 2: M = xw20 + Lu@xw21 ; out += softmax(...) @ M
    k_gemm_part   <<<gpart,    256, 0, stream>>>(Lu, xw21T, Yp);
    k_reduce_bf16T<<<NF / 256, 256, 0, stream>>>(Yp, xw20, MT);
    k_attn_part   <<<gpart,    256, 0, stream>>>(Lu, er2, ec2, MT, Yp, mp, sp);
    k_attn_reduce <<<NN / 8,   256, 0, stream>>>(Yp, mp, sp, out);
}
